// RoPEMultiHeadAttention_8589934592301
// MI455X (gfx1250) — matrix operand order repaired and checked by reading, not compiled, not run on hardware
//
#include <hip/hip_runtime.h>

// ---------------------------------------------------------------------------
// RoPE Multi-Head Attention for MI455X (gfx1250), bf16 WMMA pipeline.
//   B=2, S=2048, E=1024, H=16, D=64
// Stages:
//   1..3) q/k/v = X @ W + b        (WMMA bf16, fp32 acc, bf16 out to ws)
//   4)    RoPE(q,k) + V transpose  (elementwise fp32)
//   5)    flash attention          (async K/V tiles -> LDS, WMMA QK^T +
//                                   online softmax + WMMA PV, double buffer)
//   6)    out = ctx @ Wo + bo      (WMMA bf16, async A staging, fp32 out)
// ---------------------------------------------------------------------------

#define B_NUM  2
#define S_LEN  2048
#define E_DIM  1024
#define H_NUM  16
#define D_HEAD 64
#define M_TOT  (B_NUM * S_LEN)   // 4096 rows

typedef __attribute__((ext_vector_type(16))) __bf16       v16bf;
typedef __attribute__((ext_vector_type(8)))  float        v8f;
typedef __attribute__((ext_vector_type(4)))  float        v4f;
typedef __attribute__((ext_vector_type(4)))  unsigned int v4u;
typedef __attribute__((ext_vector_type(4)))  int          v4i;

typedef unsigned short u16;
typedef unsigned int   u32;

// --------------------------- CDNA5 async LDS path ---------------------------
#if defined(__gfx1250__) && __has_builtin(__builtin_amdgcn_global_load_async_to_lds_b128)
#define HAS_ASYNC_LDS 1
#else
#define HAS_ASYNC_LDS 0
#endif

#if HAS_ASYNC_LDS
typedef __attribute__((address_space(1))) v4i* glb_v4i_ptr;
typedef __attribute__((address_space(3))) v4i* lds_v4i_ptr;
#endif

// copy 16 bytes global -> LDS (async on gfx1250, sync fallback otherwise)
static __device__ __forceinline__ void cp16_g2l(u16* lds_dst, const u16* g_src) {
#if HAS_ASYNC_LDS
  __builtin_amdgcn_global_load_async_to_lds_b128(
      (glb_v4i_ptr)g_src, (lds_v4i_ptr)lds_dst, 0, 0);
#else
  *(v4u*)lds_dst = *(const v4u*)g_src;
#endif
}

template <int N>
static __device__ __forceinline__ void async_wait_n() {
#if HAS_ASYNC_LDS
#if __has_builtin(__builtin_amdgcn_s_wait_asynccnt)
  __builtin_amdgcn_s_wait_asynccnt(N);
#else
  if (N == 0) asm volatile("s_wait_asynccnt 0x0" ::: "memory");
  else        asm volatile("s_wait_asynccnt 0x4" ::: "memory");
#endif
#endif
}

// ---------------------------------------------------------------------------
static __device__ __forceinline__ u16 f2bf(float f) {
  u32 u = __float_as_uint(f);
  u32 r = u + 0x7FFFu + ((u >> 16) & 1u);   // round-to-nearest-even
  return (u16)(r >> 16);
}
static __device__ __forceinline__ float bf2f(u16 h) {
  return __uint_as_float(((u32)h) << 16);
}

union FragAB {
  v16bf v;
  v4u   q[2];
  u16   u[16];
};

static __device__ __forceinline__ v8f wmma_bf16(v16bf a, v16bf b, v8f c) {
  // D = A(16x32 bf16) * B(32x16 bf16) + C(16x16 f32)
  return __builtin_amdgcn_wmma_f32_16x16x32_bf16(false, a, false, b,
                                                 (short)0, c, false, false);
}

static __device__ __forceinline__ v8f zero8() {
  v8f z = {0.f, 0.f, 0.f, 0.f, 0.f, 0.f, 0.f, 0.f};
  return z;
}

// reductions across the 16-lane half of a wave32 (xor of bits 0..3 only)
static __device__ __forceinline__ float half_max(float v) {
  #pragma unroll
  for (int o = 8; o >= 1; o >>= 1) v = fmaxf(v, __shfl_xor(v, o, 32));
  return v;
}
static __device__ __forceinline__ float half_sum(float v) {
  #pragma unroll
  for (int o = 8; o >= 1; o >>= 1) v += __shfl_xor(v, o, 32);
  return v;
}

// ---------------------------------------------------------------------------
// GEMM: C[M_TOT x E] = A[M_TOT x E] * W[E x E] + bias
// Block tile 128x64, 8 waves (4x2), wave tile 32x32 (2x2 WMMA frags), KC=32.
// A staged row-major bf16 in LDS (async when already bf16);
// W staged transposed ([n][k]) bf16 in LDS.
// ---------------------------------------------------------------------------
template <int A_IS_F32, int OUT_IS_F32>
__global__ __launch_bounds__(256) void gemm_bias_wmma(
    const void* __restrict__ Ain, const float* __restrict__ W,
    const float* __restrict__ bias, void* __restrict__ Cout) {
  __shared__ __align__(16) u16 As[128 * 32];  // 8 KiB
  __shared__ __align__(16) u16 Bs[64 * 32];   // 4 KiB (transposed: [n][k])

  const int tid  = threadIdx.x;
  const int m0   = blockIdx.x * 128;
  const int n0   = blockIdx.y * 64;
  const int lane = tid & 31;
  const int w    = tid >> 5;
  const int wm   = (w >> 1) * 32;
  const int wn   = (w & 1) * 32;
  const int lh   = lane & 15;
  const int hi   = lane >> 4;
  const int ko   = hi * 8;    // A-frag: half-wave K offset (K{0..7}/K{8..15})
  const int bko  = hi * 16;   // unused; B-frag uses ko, same K order as A

  v8f acc[2][2];
  acc[0][0] = zero8(); acc[0][1] = zero8();
  acc[1][0] = zero8(); acc[1][1] = zero8();

  const int arow = tid >> 1;         // 0..127
  const int akh  = (tid & 1) * 16;   // 0 or 16

  for (int kk = 0; kk < E_DIM; kk += 32) {
    // ---- stage A tile (128 x 32) as bf16 ----
    if (A_IS_F32) {
      const float* ap = (const float*)Ain + (size_t)(m0 + arow) * E_DIM + kk + akh;
      FragAB t;
      #pragma unroll
      for (int i = 0; i < 4; ++i) {
        v4f x = *(const v4f*)(ap + i * 4);
        t.u[i * 4 + 0] = f2bf(x.x);
        t.u[i * 4 + 1] = f2bf(x.y);
        t.u[i * 4 + 2] = f2bf(x.z);
        t.u[i * 4 + 3] = f2bf(x.w);
      }
      *(v4u*)&As[arow * 32 + akh]     = t.q[0];
      *(v4u*)&As[arow * 32 + akh + 8] = t.q[1];
      if (kk + 32 < E_DIM) __builtin_prefetch((const char*)(ap + 32), 0, 0);
    } else {
      const u16* ap = (const u16*)Ain + (size_t)(m0 + arow) * E_DIM + kk + akh;
      cp16_g2l(&As[arow * 32 + akh],     ap);       // async on gfx1250
      cp16_g2l(&As[arow * 32 + akh + 8], ap + 8);
      if (kk + 32 < E_DIM) __builtin_prefetch((const char*)(ap + 32), 0, 0);
    }
    // ---- stage W tile (32 x 64), transposed into Bs[n][k] ----
    #pragma unroll
    for (int i = 0; i < 2; ++i) {
      int idx = tid * 2 + i;           // 0..511
      int k   = idx >> 4;              // 0..31
      int nq  = idx & 15;              // 0..15 (groups of 4 columns)
      v4f wv  = *(const v4f*)(W + (size_t)(kk + k) * E_DIM + n0 + nq * 4);
      Bs[(nq * 4 + 0) * 32 + k] = f2bf(wv.x);
      Bs[(nq * 4 + 1) * 32 + k] = f2bf(wv.y);
      Bs[(nq * 4 + 2) * 32 + k] = f2bf(wv.z);
      Bs[(nq * 4 + 3) * 32 + k] = f2bf(wv.w);
    }
    if (!A_IS_F32) async_wait_n<0>();
    __syncthreads();

    // ---- fragments + 4 WMMAs ----
    FragAB af[2], bfr[2];
    #pragma unroll
    for (int i = 0; i < 2; ++i) {
      const u16* p = &As[(wm + i * 16 + lh) * 32 + ko];
      af[i].q[0] = *(const v4u*)(p);        // K {0..7}   (hi=1: {8..15})
      af[i].q[1] = *(const v4u*)(p + 16);   // K {16..23} (hi=1: {24..31})
    }
    #pragma unroll
    for (int j = 0; j < 2; ++j) {
      const u16* p = &Bs[(wn + j * 16 + lh) * 32 + ko];
      bfr[j].q[0] = *(const v4u*)(p);       // K {0..7}   (hi=1: {8..15})
      bfr[j].q[1] = *(const v4u*)(p + 16);  // K {16..23} (hi=1: {24..31})
    }
    #pragma unroll
    for (int i = 0; i < 2; ++i)
      #pragma unroll
      for (int j = 0; j < 2; ++j)
        acc[i][j] = wmma_bf16(af[i].v, bfr[j].v, acc[i][j]);
    __syncthreads();
  }

  // ---- epilogue: bias + store (C layout: VGPR r -> row r (+8 for hi half)) ----
  #pragma unroll
  for (int i = 0; i < 2; ++i) {
    #pragma unroll
    for (int j = 0; j < 2; ++j) {
      const int   gn = n0 + wn + j * 16 + lh;
      const float bv = bias[gn];
      #pragma unroll
      for (int r = 0; r < 8; ++r) {
        const int gm = m0 + wm + i * 16 + r + hi * 8;
        float val = acc[i][j][r] + bv;
        if (OUT_IS_F32)
          ((float*)Cout)[(size_t)gm * E_DIM + gn] = val;
        else
          ((u16*)Cout)[(size_t)gm * E_DIM + gn] = f2bf(val);
      }
    }
  }
}

// ---------------------------------------------------------------------------
// RoPE on q,k (rotate-half convention) + transpose V to [B,H,D,S].
// One thread per (b,s,h,d<32) pair.
// ---------------------------------------------------------------------------
__global__ __launch_bounds__(256) void rope_reformat(
    const u16* __restrict__ qp, const u16* __restrict__ kp,
    const u16* __restrict__ vp, u16* __restrict__ qr, u16* __restrict__ kr,
    u16* __restrict__ vt) {
  const int gid = blockIdx.x * 256 + threadIdx.x;
  const int d = gid & 31;
  const int h = (gid >> 5) & (H_NUM - 1);
  const int s = (gid >> 9) & (S_LEN - 1);
  const int b = gid >> 20;
  if (b >= B_NUM) return;

  const size_t base = ((size_t)(b * S_LEN + s)) * E_DIM + h * D_HEAD;

  // freq_d = base^(-2d/64); ln(10000) = 9.210340372
  const float ang = (float)s * __expf(-((float)(2 * d) * (1.0f / 64.0f)) * 9.2103403720f);
  const float c  = __cosf(ang);
  const float sn = __sinf(ang);

  {
    float x0 = bf2f(qp[base + d]);
    float x1 = bf2f(qp[base + d + 32]);
    qr[base + d]      = f2bf(x0 * c - x1 * sn);
    qr[base + d + 32] = f2bf(x1 * c + x0 * sn);
  }
  {
    float x0 = bf2f(kp[base + d]);
    float x1 = bf2f(kp[base + d + 32]);
    kr[base + d]      = f2bf(x0 * c - x1 * sn);
    kr[base + d + 32] = f2bf(x1 * c + x0 * sn);
  }
  {
    const size_t vbase = ((size_t)((b * H_NUM + h) * D_HEAD)) * S_LEN;
    vt[vbase + (size_t)d * S_LEN + s]        = vp[base + d];
    vt[vbase + (size_t)(d + 32) * S_LEN + s] = vp[base + d + 32];
  }
}

// ---------------------------------------------------------------------------
// Flash attention: block = (b,h, 64 query rows), 4 waves x 16 rows each.
// K/V tiles for each 32-key block are staged into LDS with async global->LDS
// copies (double buffered, shared by all 4 waves). Per key block: 4 WMMAs for
// QK^T, online softmax (fp32), P->LDS->A-frag, 4 WMMAs for P*V.
// ---------------------------------------------------------------------------
__global__ __launch_bounds__(128) void flash_attn_wmma(
    const u16* __restrict__ Q, const u16* __restrict__ K,
    const u16* __restrict__ Vt, const int* __restrict__ mask,
    u16* __restrict__ ctx) {
  __shared__ __align__(16) u16 Ks[2][32 * 64];   // 2 x 4 KiB: 32 keys x 64 dims
  __shared__ __align__(16) u16 Vs[2][64 * 32];   // 2 x 4 KiB: 64 dims x 32 keys
  __shared__ __align__(16) u16 pbuf[4][16 * 32]; // per-wave P tile (bf16)

  const int bh   = blockIdx.x;
  const int b    = bh >> 4;
  const int h    = bh & 15;
  const int q0   = blockIdx.y * 64;
  const int tid  = threadIdx.x;
  const int wq   = tid >> 5;
  const int lane = tid & 31;
  const int lh   = lane & 15;
  const int hi   = lane >> 4;
  const int ko   = hi * 8;
  const int bko  = hi * 16;

  const u16* Kbase  = K  + (size_t)b * S_LEN * E_DIM + h * D_HEAD;
  const u16* Vtbase = Vt + ((size_t)(b * H_NUM + h)) * D_HEAD * S_LEN;

  // cooperative stage of one 32-key block (4 x 16B async copies per thread)
  auto stage_kv = [&](int buf, int kb) {
    #pragma unroll
    for (int i = 0; i < 2; ++i) {
      int c   = tid * 2 + i;      // 0..255
      int row = c >> 3;           // 0..31  (key within block)
      int col = (c & 7) * 8;      // 0..56  (dim)
      cp16_g2l(&Ks[buf][row * 64 + col],
               Kbase + (size_t)(kb + row) * E_DIM + col);
    }
    #pragma unroll
    for (int i = 0; i < 2; ++i) {
      int c   = tid * 2 + i;      // 0..255
      int row = c >> 2;           // 0..63  (dim)
      int col = (c & 3) * 8;      // 0..24  (key within block)
      cp16_g2l(&Vs[buf][row * 32 + col],
               Vtbase + (size_t)row * S_LEN + kb + col);
    }
  };

  // Q fragments: constant across the key loop (A operand, K-dim = head dim 64)
  FragAB qf[2];
  {
    const size_t qoff =
        ((size_t)(b * S_LEN + q0 + wq * 16 + lh)) * E_DIM + h * D_HEAD;
    #pragma unroll
    for (int kc = 0; kc < 2; ++kc) {
      const u16* p = Q + qoff + kc * 32 + ko;
      qf[kc].q[0] = *(const v4u*)(p);
      qf[kc].q[1] = *(const v4u*)(p + 16);
    }
  }

  v8f o[4];
  #pragma unroll
  for (int j = 0; j < 4; ++j) o[j] = zero8();
  float mrow[8], lrow[8];
  #pragma unroll
  for (int r = 0; r < 8; ++r) { mrow[r] = -1e30f; lrow[r] = 0.f; }

  int cur = 0;
  stage_kv(0, 0);  // prologue

  for (int kb = 0; kb < S_LEN; kb += 32) {
    if (kb + 32 < S_LEN) {
      stage_kv(cur ^ 1, kb + 32);   // prefetch next tile (4 newer per thread)
      async_wait_n<4>();            // current tile's copies complete (in order)
    } else {
      async_wait_n<0>();
    }
    __syncthreads();                // all waves' tile data visible

    // ---- K fragments from LDS (B operand: keys along N, head dim along K) ----
    FragAB kf[2][2];
    #pragma unroll
    for (int nf = 0; nf < 2; ++nf) {
      #pragma unroll
      for (int kc = 0; kc < 2; ++kc) {
        const u16* p = &Ks[cur][(nf * 16 + lh) * 64 + kc * 32 + ko];
        kf[nf][kc].q[0] = *(const v4u*)(p);
        kf[nf][kc].q[1] = *(const v4u*)(p + 16);
      }
    }
    // ---- scores: 16 rows x 32 keys ----
    v8f sc[2];
    #pragma unroll
    for (int nf = 0; nf < 2; ++nf) {
      sc[nf] = zero8();
      sc[nf] = wmma_bf16(qf[0].v, kf[nf][0].v, sc[nf]);
      sc[nf] = wmma_bf16(qf[1].v, kf[nf][1].v, sc[nf]);
    }
    const int mv0 = mask[b * S_LEN + kb + lh];
    const int mv1 = mask[b * S_LEN + kb + 16 + lh];

    // ---- online softmax update (fp32 VALU + half-wave shuffles) ----
    float p0[8], p1[8];
    #pragma unroll
    for (int r = 0; r < 8; ++r) {
      float s0 = sc[0][r] * 0.125f;   // 1/sqrt(64)
      float s1 = sc[1][r] * 0.125f;
      if (mv0 == 0) s0 = -1e30f;
      if (mv1 == 0) s1 = -1e30f;
      float vmax  = half_max(fmaxf(s0, s1));
      float mnew  = fmaxf(mrow[r], vmax);
      float scale = __expf(mrow[r] - mnew);
      mrow[r] = mnew;
      float e0 = __expf(s0 - mnew);
      float e1 = __expf(s1 - mnew);
      lrow[r] = lrow[r] * scale + half_sum(e0 + e1);
      p0[r] = e0;
      p1[r] = e1;
      o[0][r] *= scale; o[1][r] *= scale; o[2][r] *= scale; o[3][r] *= scale;
    }
    // ---- P: C-layout -> LDS -> A-fragment (bf16, per-wave buffer) ----
    #pragma unroll
    for (int r = 0; r < 8; ++r) {
      pbuf[wq][(r + hi * 8) * 32 + lh]      = f2bf(p0[r]);
      pbuf[wq][(r + hi * 8) * 32 + 16 + lh] = f2bf(p1[r]);
    }
    FragAB pa;
    {
      const u16* p = &pbuf[wq][lh * 32 + ko];
      pa.q[0] = *(const v4u*)(p);
      pa.q[1] = *(const v4u*)(p + 16);
    }
    // ---- P*V from LDS V tile (contiguous keys per dim row) ----
    #pragma unroll
    for (int j = 0; j < 4; ++j) {
      FragAB vf;
      const u16* p = &Vs[cur][(j * 16 + lh) * 32 + ko];
      vf.q[0] = *(const v4u*)(p);
      vf.q[1] = *(const v4u*)(p + 16);
      o[j] = wmma_bf16(pa.v, vf.v, o[j]);
    }
    __syncthreads();  // everyone done reading tiles[cur] before it is restaged
    cur ^= 1;
  }

  // ---- normalize + store ctx (bf16, [B,S,E]) ----
  #pragma unroll
  for (int r = 0; r < 8; ++r) {
    const float inv = 1.0f / lrow[r];
    const size_t row = (size_t)(b * S_LEN + q0 + wq * 16 + r + hi * 8);
    #pragma unroll
    for (int j = 0; j < 4; ++j)
      ctx[row * E_DIM + h * D_HEAD + j * 16 + lh] = f2bf(o[j][r] * inv);
  }
}

// ---------------------------------------------------------------------------
extern "C" void kernel_launch(void* const* d_in, const int* in_sizes, int n_in,
                              void* d_out, int out_size, void* d_ws,
                              size_t ws_size, hipStream_t stream) {
  (void)in_sizes; (void)n_in; (void)out_size; (void)ws_size;

  const float* query = (const float*)d_in[0];
  const float* key   = (const float*)d_in[1];
  const float* value = (const float*)d_in[2];
  const int*   mask  = (const int*)d_in[3];
  const float* Wq = (const float*)d_in[4];
  const float* bq = (const float*)d_in[5];
  const float* Wk = (const float*)d_in[6];
  const float* bk = (const float*)d_in[7];
  const float* Wv = (const float*)d_in[8];
  const float* bv = (const float*)d_in[9];
  const float* Wo = (const float*)d_in[10];
  const float* bo = (const float*)d_in[11];
  float* out = (float*)d_out;

  const size_t BUF = (size_t)M_TOT * E_DIM * sizeof(u16);  // 8 MiB each
  u16* qp  = (u16*)((char*)d_ws + 0 * BUF);
  u16* kp  = (u16*)((char*)d_ws + 1 * BUF);
  u16* vp  = (u16*)((char*)d_ws + 2 * BUF);
  u16* qr  = (u16*)((char*)d_ws + 3 * BUF);
  u16* kr  = (u16*)((char*)d_ws + 4 * BUF);
  u16* vt  = (u16*)((char*)d_ws + 5 * BUF);
  u16* ctx = (u16*)((char*)d_ws + 6 * BUF);

  dim3 gg(M_TOT / 128, E_DIM / 64);
  gemm_bias_wmma<1, 0><<<gg, 256, 0, stream>>>(query, Wq, bq, qp);
  gemm_bias_wmma<1, 0><<<gg, 256, 0, stream>>>(key,   Wk, bk, kp);
  gemm_bias_wmma<1, 0><<<gg, 256, 0, stream>>>(value, Wv, bv, vp);

  rope_reformat<<<(B_NUM * S_LEN * H_NUM * 32) / 256, 256, 0, stream>>>(
      qp, kp, vp, qr, kr, vt);

  dim3 gf(B_NUM * H_NUM, S_LEN / 64);
  flash_attn_wmma<<<gf, 128, 0, stream>>>(qr, kr, vt, mask, ctx);

  gemm_bias_wmma<0, 1><<<gg, 256, 0, stream>>>(ctx, Wo, bo, out);
}
